// LSTM_90666759618886
// MI455X (gfx1250) — compile-verified, matrix operand order checked by reading, not run on hardware
//
#include <hip/hip_runtime.h>
#include <hip/hip_bf16.h>
#include <stdint.h>

// ---------------------------------------------------------------------------
// LSTM (T=32768, H=512) for MI455X / gfx1250.
//
// 16 persistent workgroups, each owns 32 hidden elements = 128 gate rows of
// W_hh. Each wave holds its 8 WMMA A-fragments (f16 weights) in VGPRs for the
// whole run. The shared h vector is published to global memory as f16 and
// pulled into LDS each step with one Tensor-Data-Mover tensor_load_to_lds
// (async, s_wait_tensorcnt). B fragments are 16 pipelined ds_load_b128
// (sched_group_barrier keeps all DS reads ahead of the 8 back-to-back
// accumulating WMMAs, so ~one LDS latency is exposed per step). All 16 B
// columns carry the same h chunk, so every column of D equals the mat-vec and
// no lane masking is needed. Cell state c and the input/bias rows stay in
// registers of the updater wave. Cross-WGP sync: agent-scope generation
// barrier.
// ---------------------------------------------------------------------------

typedef _Float16 v8h  __attribute__((ext_vector_type(8)));
typedef _Float16 v16h __attribute__((ext_vector_type(16)));
typedef float    v8f  __attribute__((ext_vector_type(8)));

#define NWG   16      // workgroups (one per WGP slice)
#define BLK   512     // threads per block = 16 waves
#define HID   512
#define ROWS  128     // gate rows per WGP (32 each of i,f,g,o)

// dynamic LDS layout (bytes)
#define OFF_H16   0        // 512 f16  (h vector, 1KB)
#define OFF_PART  1024     // 2*128 f32 partial gate sums
#define SMEM_BYTES 2048

__device__ __forceinline__ float sigmoidf_(float x) {
    return 1.0f / (1.0f + __expf(-x));
}

__device__ __forceinline__ void grid_sync(unsigned* cnt, unsigned* gen,
                                          unsigned target) {
    __syncthreads();
    if (threadIdx.x == 0) {
        unsigned arrived = __hip_atomic_fetch_add(cnt, 1u, __ATOMIC_ACQ_REL,
                                                  __HIP_MEMORY_SCOPE_AGENT);
        if (arrived == NWG - 1) {
            __hip_atomic_store(cnt, 0u, __ATOMIC_RELAXED,
                               __HIP_MEMORY_SCOPE_AGENT);
            __hip_atomic_store(gen, target, __ATOMIC_RELEASE,
                               __HIP_MEMORY_SCOPE_AGENT);
        } else {
            while (__hip_atomic_load(gen, __ATOMIC_ACQUIRE,
                                     __HIP_MEMORY_SCOPE_AGENT) < target) {
                __builtin_amdgcn_s_sleep(1);
            }
        }
    }
    __syncthreads();
}

__global__ void lstm_init_kernel(const float* __restrict__ h0,
                                 unsigned short* __restrict__ hbuf16,
                                 unsigned* __restrict__ bar) {
    int t = threadIdx.x;
    if (t < 2) bar[t] = 0u;
    if (t < HID) {
        _Float16 hv = (_Float16)h0[t];
        hbuf16[t] = __builtin_bit_cast(unsigned short, hv);
    }
}

__global__ __launch_bounds__(BLK, 1)
void lstm_main_kernel(const float* __restrict__ input_seq,
                      const float* __restrict__ c0,
                      const float* __restrict__ W_ih,
                      const float* __restrict__ W_hh,
                      const float* __restrict__ b_ih,
                      const float* __restrict__ b_hh,
                      const float* __restrict__ W_lin,
                      const float* __restrict__ b_lin,
                      unsigned short* __restrict__ hbuf16,
                      unsigned* __restrict__ bar,
                      float* __restrict__ out,
                      int T) {
    extern __shared__ char smem[];
    _Float16* h16  = (_Float16*)(smem + OFF_H16);
    float*    part = (float*)(smem + OFF_PART);

    const int tid  = threadIdx.x;
    const int wg   = blockIdx.x;
    const int lane = tid & 31;
    const int wv   = tid >> 5;
    const int mt   = wv & 7;      // which 16-row tile this wave accumulates
    const int kh   = wv >> 3;     // which half of K (8 ktiles each)
    const int half = lane >> 4;   // half-wave (selects K sub-range of frag)

    // ---- one-time init: load this wave's 8 A-fragments into registers ----
    // Fragment layout (16-bit A, 16x32): lane m holds row m; element e maps to
    // K = (e<8 ? e : e+8) + half*8, i.e. e0..7 -> K contiguous at half*8,
    // e8..15 -> K contiguous at 16+half*8.
    const int m    = lane & 15;
    const int r0   = mt * 16 + m;          // slice row for this lane
    const int gate = r0 >> 5;
    const int j0   = wg * 32 + (r0 & 31);
    const float* wrow = W_hh + (gate * HID + j0) * HID;

    v16h a_frag[8];
#pragma unroll
    for (int kk = 0; kk < 8; ++kk) {
        int k0 = (kh * 8 + kk) * 32 + half * 8;
        v16h a;
#pragma unroll
        for (int e = 0; e < 8; ++e) a[e] = (_Float16)wrow[k0 + e];
#pragma unroll
        for (int e = 0; e < 8; ++e) a[e + 8] = (_Float16)wrow[k0 + 16 + e];
        a_frag[kk] = a;
    }

    // updater wave keeps its input-weight & bias rows + cell state in regs
    float wih_r[4], bias_r[4];
    float creg = 0.0f;
    if (tid < 32) {
#pragma unroll
        for (int g = 0; g < 4; ++g) {
            int gr = g * HID + wg * 32 + tid;
            wih_r[g]  = W_ih[gr];
            bias_r[g] = b_ih[gr] + b_hh[gr];
        }
        creg = c0[wg * 32 + tid];
    }
    __syncthreads();

    const uint64_t ga  = (uint64_t)hbuf16;
    const unsigned lds = (unsigned)(uintptr_t)h16;

    for (int t = 0; t < T; ++t) {
        // ------ 1. pull shared h_t (512 f16, 1KB) from global into LDS ------
#if __has_builtin(__builtin_amdgcn_tensor_load_to_lds)
        if (tid < 32) {   // one wave issues the tensor DMA
            typedef unsigned int u32x4 __attribute__((ext_vector_type(4)));
            typedef int i32x4 __attribute__((ext_vector_type(4)));
            typedef int i32x8 __attribute__((ext_vector_type(8)));
            u32x4 g0;
            g0[0] = 1u;                                      // count=1
            g0[1] = lds;                                     // lds_addr
            g0[2] = (unsigned)(ga & 0xFFFFFFFFu);            // global lo
            g0[3] = (unsigned)((ga >> 32) & 0x1FFFFFFu)      // global hi
                    | (2u << 30);                            // type=2 (image)
            i32x8 g1;
            g1[0] = (int)(1u << 16);        // data_size = 2B (f16)
            g1[1] = (int)(512u << 16);      // tensor_dim0[15:0] = 512
            g1[2] = (int)(1u << 16);        // tensor_dim1 = 1
            g1[3] = (int)(512u << 16);      // tile_dim0 = 512
            g1[4] = 1;                      // tile_dim1 = 1
            g1[5] = 512;                    // tensor_dim0_stride = 512
            g1[6] = 0;
            g1[7] = 0;
            i32x4 z4 = {0, 0, 0, 0};
            i32x8 z8 = {0, 0, 0, 0, 0, 0, 0, 0};
            __builtin_amdgcn_tensor_load_to_lds(g0, g1, z4, z4, z8, 0);
            __builtin_amdgcn_s_wait_tensorcnt(0);
        }
        __syncthreads();
#else
        if (tid < HID) {
            unsigned short us = __hip_atomic_load(&hbuf16[tid], __ATOMIC_ACQUIRE,
                                                  __HIP_MEMORY_SCOPE_AGENT);
            h16[tid] = __builtin_bit_cast(_Float16, us);
        }
        __syncthreads();
#endif

        // ------ 2. gate mat-vec via WMMA: 8 ktiles/wave, A in registers ------
        v16h b_frag[8];
#pragma unroll
        for (int kk = 0; kk < 8; ++kk) {
            int kbase = (kh * 8 + kk) * 32 + half * 8;
            v8h b0 = *(const v8h*)(h16 + kbase);        // K: +0..7
            v8h b1 = *(const v8h*)(h16 + kbase + 16);   // K: +16..23
            b_frag[kk] = __builtin_shufflevector(b0, b1, 0, 1, 2, 3, 4, 5, 6,
                                                 7, 8, 9, 10, 11, 12, 13, 14,
                                                 15);
        }
        v8f acc = {};
#pragma unroll
        for (int kk = 0; kk < 8; ++kk) {
            acc = __builtin_amdgcn_wmma_f32_16x16x32_f16(
                false, a_frag[kk], false, b_frag[kk], (short)0, acc, false,
                false);
        }
#if __has_builtin(__builtin_amdgcn_sched_group_barrier)
        // keep the 16 ds_load_b128 issued ahead of the 8 WMMAs so the loads
        // pipeline and only ~one LDS latency is exposed per step
        __builtin_amdgcn_sched_group_barrier(0x100, 16, 0);  // DS reads
        __builtin_amdgcn_sched_group_barrier(0x008, 8, 0);   // WMMA
#endif
        // column 0 of D: lane 0 has rows mt*16+0..7, lane 16 rows +8..15
        if ((lane & 15) == 0) {
            *(v8f*)&part[kh * ROWS + mt * 16 + half * 8] = acc;
        }
        __syncthreads();

        // ------ 3. cell update for our 32 hidden elements ------
        if (tid < 32) {
            float x = input_seq[t];
            float gv[4];
#pragma unroll
            for (int g = 0; g < 4; ++g) {
                int r = g * 32 + tid;
                gv[g] = part[r] + part[ROWS + r] + x * wih_r[g] + bias_r[g];
            }
            float i_ = sigmoidf_(gv[0]);
            float f_ = sigmoidf_(gv[1]);
            float g_ = tanhf(gv[2]);
            float o_ = sigmoidf_(gv[3]);
            creg = f_ * creg + i_ * g_;
            float hn = o_ * tanhf(creg);
            unsigned short us =
                __builtin_bit_cast(unsigned short, (_Float16)hn);
            __hip_atomic_store(&hbuf16[wg * 32 + tid], us, __ATOMIC_RELEASE,
                               __HIP_MEMORY_SCOPE_AGENT);
        }

        // ------ 4. all 16 WGPs rendezvous before next timestep ------
        grid_sync(&bar[0], &bar[1], (unsigned)(t + 1));
    }

    // ------ final projection: out = W_lin . h_T + b_lin (wg 0 only) ------
    if (wg == 0 && tid < 32) {
        float s = 0.0f;
#pragma unroll
        for (int k2 = 0; k2 < 16; ++k2) {
            int j = tid + 32 * k2;
            unsigned short us = __hip_atomic_load(&hbuf16[j], __ATOMIC_ACQUIRE,
                                                  __HIP_MEMORY_SCOPE_AGENT);
            s += W_lin[j] * (float)__builtin_bit_cast(_Float16, us);
        }
#pragma unroll
        for (int off = 16; off > 0; off >>= 1) s += __shfl_down(s, off, 32);
        if (tid == 0) out[0] = s + b_lin[0];
    }
}

extern "C" void kernel_launch(void* const* d_in, const int* in_sizes, int n_in,
                              void* d_out, int out_size, void* d_ws, size_t ws_size,
                              hipStream_t stream) {
    const float* input_seq = (const float*)d_in[0];
    const float* h0        = (const float*)d_in[1];
    const float* c0        = (const float*)d_in[2];
    const float* W_ih      = (const float*)d_in[3];
    const float* W_hh      = (const float*)d_in[4];
    const float* b_ih      = (const float*)d_in[5];
    const float* b_hh      = (const float*)d_in[6];
    const float* W_lin     = (const float*)d_in[7];
    const float* b_lin     = (const float*)d_in[8];
    float* out = (float*)d_out;
    const int T = in_sizes[0];

    // workspace: [0..255] barrier (count, generation), [256..] f16 h vector
    unsigned* bar          = (unsigned*)d_ws;
    unsigned short* hbuf16 = (unsigned short*)((char*)d_ws + 256);

    lstm_init_kernel<<<1, BLK, 0, stream>>>(h0, hbuf16, bar);
    lstm_main_kernel<<<NWG, BLK, SMEM_BYTES, stream>>>(
        input_seq, c0, W_ih, W_hh, b_ih, b_hh, W_lin, b_lin,
        hbuf16, bar, out, T);
}
